// Attention_36120674959366
// MI455X (gfx1250) — matrix operand order repaired and checked by reading, not compiled, not run on hardware
//
#include <hip/hip_runtime.h>
#include <hip/hip_bf16.h>

#define B_      2
#define S_      2048
#define DIM_    2048
#define NH_     16
#define NKVH_   4
#define HD_     128
#define BS_     (B_ * S_)
#define SCALE_  0.088388347648318447f   // 1/sqrt(128)
#define LN_EPS_ 1e-5f

typedef __attribute__((ext_vector_type(16))) __bf16 v16bf;
typedef __attribute__((ext_vector_type(8)))  float  v8f;
typedef __attribute__((ext_vector_type(4)))  unsigned int u32x4;
typedef __attribute__((ext_vector_type(8)))  int          i32x8;
typedef __attribute__((ext_vector_type(4)))  int          i32x4;

#if defined(__has_builtin)
#if __has_builtin(__builtin_amdgcn_tensor_load_to_lds) && \
    __has_builtin(__builtin_amdgcn_s_wait_tensorcnt)
#define HAVE_TDM 1
#endif
#endif
#ifndef HAVE_TDM
#define HAVE_TDM 0
#endif

__device__ inline __bf16 tobf(float f) {
  unsigned u = __builtin_bit_cast(unsigned, f);
  unsigned r = (u + 0x7FFFu + ((u >> 16) & 1u)) >> 16;   // round-to-nearest-even
  return __builtin_bit_cast(__bf16, (unsigned short)r);
}

__device__ inline v8f wmma_bf16(v16bf a, v16bf b, v8f c) {
  // D(f32 16x16) = A(bf16 16x32) * B(bf16 32x16) + C
  return __builtin_amdgcn_wmma_f32_16x16x32_bf16(false, a, false, b, (short)0, c, false, false);
}

// K-index inside a 16x32 bf16 A-fragment for element e of this lane (ISA 7.12.2).
__device__ inline int amap(int e, int half) {
  int p = e >> 1, j = e & 1;
  return ((p & 4) << 2) + (half << 3) + ((p & 3) << 1) + j;
}

// ------------------------------------------------------------------ TDM -----
// 2D tile DMA: global (tile1 rows x tile0 bf16 elems, row stride stride0 elems)
// -> LDS at lds_addr, optional LDS pad per row via padcfg (D# group1 bits).
// Descriptor layout per CDNA5 ISA.
#if HAVE_TDM
__device__ inline void tdm_load_2d(unsigned lds_addr, const void* g,
                                   unsigned tile0, unsigned tile1,
                                   unsigned long long stride0, unsigned padcfg) {
  unsigned long long ga = (unsigned long long)g;
  u32x4 g0;
  g0[0] = 1u;                                            // count=1 (valid user D#)
  g0[1] = lds_addr;                                      // LDS byte address
  g0[2] = (unsigned)ga;                                  // global_addr[31:0]
  g0[3] = (unsigned)((ga >> 32) & 0x1FFFFFFu) | 0x80000000u; // ga[56:32] | type=2
  i32x8 g1;
  g1[0] = (int)(0x10000u | padcfg);                      // data_size=1 (2B) | pad cfg
  g1[1] = (int)((tile0 & 0xFFFFu) << 16);                // tensor_dim0[15:0]
  g1[2] = (int)((tile1 & 0xFFFFu) << 16);                // tensor_dim1[15:0]
  g1[3] = (int)((tile0 & 0xFFFFu) << 16);                // tile_dim0
  g1[4] = (int)(tile1 & 0xFFFFu);                        // tile_dim1 (tile_dim2=0)
  g1[5] = (int)(stride0 & 0xFFFFFFFFull);                // tensor_dim0_stride lo
  g1[6] = (int)((stride0 >> 32) & 0xFFFFull);            // stride hi (dim1 stride 0)
  g1[7] = 0;
  i32x4 z = {0, 0, 0, 0};
#if __clang_major__ >= 23
  i32x8 z8 = {0, 0, 0, 0, 0, 0, 0, 0};
  __builtin_amdgcn_tensor_load_to_lds(g0, g1, z, z, z8, 0);
#else
  __builtin_amdgcn_tensor_load_to_lds(g0, g1, z, z, 0);
#endif
}
#endif

// pad cfg: GEMM A-tile rows: 16 DWORDs data + 1 DWORD pad -> [128][34] bf16
#define APAD_ ((1u << 20) | (3u << 22) | (0u << 25))
// pad cfg: attn K/V rows: 64 DWORDs data + 2 DWORDs pad -> [32][132] bf16
#define KPAD_ ((1u << 20) | (5u << 22) | (1u << 25))

// ---------------------------------------------------------------- cumsum ----
__global__ void k_cumsum(const int* __restrict__ lab, int* __restrict__ cs) {
  if (threadIdx.x == 0) {
    int b = blockIdx.x, acc = 0;
    for (int i = 0; i < S_; ++i) { acc += lab[b * S_ + i]; cs[b * S_ + i] = acc; }
  }
}

// ------------------------------------------------------- layernorm + cast ---
__global__ __launch_bounds__(256) void k_ln(const float* __restrict__ X,
                                            const float* __restrict__ g,
                                            const float* __restrict__ bta,
                                            __bf16* __restrict__ out) {
  __shared__ float sh[8];
  int r = blockIdx.x, tid = threadIdx.x;
  const float* x = X + (size_t)r * DIM_;
  float v[8];
  float s = 0.f;
#pragma unroll
  for (int i = 0; i < 8; ++i) { v[i] = x[tid + i * 256]; s += v[i]; }
#pragma unroll
  for (int o = 16; o; o >>= 1) s += __shfl_xor(s, o, 32);
  if ((tid & 31) == 0) sh[tid >> 5] = s;
  __syncthreads();
  if (tid == 0) { float t = 0.f; for (int i = 0; i < 8; ++i) t += sh[i]; sh[0] = t; }
  __syncthreads();
  float mu = sh[0] / (float)DIM_;
  __syncthreads();
  float s2 = 0.f;
#pragma unroll
  for (int i = 0; i < 8; ++i) { float d = v[i] - mu; s2 += d * d; }
#pragma unroll
  for (int o = 16; o; o >>= 1) s2 += __shfl_xor(s2, o, 32);
  if ((tid & 31) == 0) sh[tid >> 5] = s2;
  __syncthreads();
  if (tid == 0) { float t = 0.f; for (int i = 0; i < 8; ++i) t += sh[i]; sh[0] = t; }
  __syncthreads();
  float rs = rsqrtf(sh[0] / (float)DIM_ + LN_EPS_);
  __bf16* o = out + (size_t)r * DIM_;
#pragma unroll
  for (int i = 0; i < 8; ++i) {
    int c = tid + i * 256;
    o[c] = tobf((v[i] - mu) * rs * g[c] + bta[c]);
  }
}

// ------------------------------------------------- GEMM: C = A(bf16) W^T ----
// A [M x K] bf16 row-major, W [N x K] f32 row-major, C [M x N] f32.
// 128x128 tile / block, 8 wave32s as 2(M) x 4(N); wave = 64x32 = 4x2 WMMA tiles.
// A-tile fills LDS via the Tensor Data Mover (wave 0 issues one descriptor),
// overlapping with the f32->bf16 staging of the W-tile by all threads.
__global__ __launch_bounds__(256) void k_gemm(const __bf16* __restrict__ A,
                                              const float* __restrict__ W,
                                              float* __restrict__ C,
                                              int N, int K) {
  __shared__ __bf16 As[128][34];
  __shared__ __bf16 Bs[32][132];
  int tid = threadIdx.x, lane = tid & 31, wv = tid >> 5;
  int wm = wv >> 2, wn = wv & 3;
  int half = lane >> 4, ln = lane & 15;
  int m0 = blockIdx.y * 128, n0 = blockIdx.x * 128;
#if HAVE_TDM
  unsigned as_lds = (unsigned)(unsigned long long)(&As[0][0]);
#endif

  v8f acc[4][2] = {};
  for (int k0 = 0; k0 < K; k0 += 32) {
    __syncthreads();
#if HAVE_TDM
    if (wv == 0)
      tdm_load_2d(as_lds, A + (size_t)m0 * K + k0, 32u, 128u,
                  (unsigned long long)K, APAD_);
#else
    for (int i = tid; i < 128 * 32; i += 256) {
      int r = i >> 5, c = i & 31;
      As[r][c] = A[(size_t)(m0 + r) * K + k0 + c];
    }
#endif
    if (k0 + 32 < K)
      __builtin_prefetch(&W[(size_t)(n0 + (tid >> 1)) * K + k0 + 32], 0, 3);
    for (int i = tid; i < 128 * 32; i += 256) {
      int n = i >> 5, c = i & 31;
      Bs[c][n] = tobf(W[(size_t)(n0 + n) * K + k0 + c]);
    }
#if HAVE_TDM
    if (wv == 0) __builtin_amdgcn_s_wait_tensorcnt(0);
#endif
    __syncthreads();

    v16bf bfr[2];
#pragma unroll
    for (int ns = 0; ns < 2; ++ns)
#pragma unroll
      for (int e = 0; e < 16; ++e)
        bfr[ns][e] = Bs[amap(e, half)][wn * 32 + ns * 16 + ln];

#pragma unroll
    for (int ms = 0; ms < 4; ++ms) {
      v16bf af;
      int row = wm * 64 + ms * 16 + ln;
#pragma unroll
      for (int e = 0; e < 16; ++e) af[e] = As[row][amap(e, half)];
      acc[ms][0] = wmma_bf16(af, bfr[0], acc[ms][0]);
      acc[ms][1] = wmma_bf16(af, bfr[1], acc[ms][1]);
    }
  }
#pragma unroll
  for (int ms = 0; ms < 4; ++ms)
#pragma unroll
    for (int ns = 0; ns < 2; ++ns)
#pragma unroll
      for (int v = 0; v < 8; ++v)
        C[(size_t)(m0 + wm * 64 + ms * 16 + v + 8 * half) * N +
          n0 + wn * 32 + ns * 16 + ln] = acc[ms][ns][v];
}

// ------------------------------------------------ RoPE + per-head packing ---
// in: [B*S][nheads*128] f32 ; out: [B*nheads][S][128] bf16
__global__ __launch_bounds__(256) void k_rope(const float* __restrict__ in,
                                              const float* __restrict__ fr,
                                              __bf16* __restrict__ out,
                                              int nheads, int dorope) {
  int idx = blockIdx.x * 256 + threadIdx.x;   // pair index
  int j = idx & 63;
  int t = idx >> 6;
  int h = t % nheads; t /= nheads;
  int s = t % S_;
  int b = t / S_;
  const float* row = in + ((size_t)(b * S_ + s)) * (nheads * 128) + h * 128;
  float xr = row[2 * j], xi = row[2 * j + 1];
  float orr, oii;
  if (dorope) {
    float c = fr[(s * 64 + j) * 2], sn = fr[(s * 64 + j) * 2 + 1];
    orr = xr * c - xi * sn;
    oii = xr * sn + xi * c;
  } else {
    orr = xr; oii = xi;
  }
  __bf16* o = out + (((size_t)(b * nheads + h)) * S_ + s) * 128 + 2 * j;
  o[0] = tobf(orr);
  o[1] = tobf(oii);
}

// ----------------------------------------------------- flash attention ------
// Q [B*NH][S][128] bf16, K/V [B*NKVH][S][128] bf16, cs [B][S] int,
// out [B*S][2048] f32.  Block = (qblock of 128 rows, b*h); wave owns 16 q rows.
// K/V tiles are DMA'd into LDS by the Tensor Data Mover (waves 0/1).
__global__ __launch_bounds__(256) void k_attn(const __bf16* __restrict__ Q,
                                              const __bf16* __restrict__ Kg,
                                              const __bf16* __restrict__ Vg,
                                              const int* __restrict__ cs,
                                              float* __restrict__ out) {
  __shared__ __bf16 Ks[32][132];
  __shared__ __bf16 Vs[32][132];
  __shared__ __bf16 Ps[8][16][36];

  int tid = threadIdx.x, lane = tid & 31, w = tid >> 5;
  int half = lane >> 4, ln = lane & 15;
  int qb = blockIdx.x, bh = blockIdx.y;
  int b = bh >> 4, h = bh & 15, kvh = h >> 2;
  int qrow0 = qb * 128 + w * 16;
  const int* csb = cs + b * S_;
#if HAVE_TDM
  unsigned ks_lds = (unsigned)(unsigned long long)(&Ks[0][0]);
  unsigned vs_lds = (unsigned)(unsigned long long)(&Vs[0][0]);
#endif

  // Q A-fragments for the 4 d-chunks of 32
  v16bf qf[4];
  const __bf16* qbase = Q + (((size_t)(b * NH_ + h)) * S_ + qrow0 + ln) * HD_;
#pragma unroll
  for (int c = 0; c < 4; ++c)
#pragma unroll
    for (int e = 0; e < 16; ++e) qf[c][e] = qbase[c * 32 + amap(e, half)];

  float m[8], l[8], csq[8];
  v8f acc[8] = {};
#pragma unroll
  for (int v = 0; v < 8; ++v) {
    m[v] = -1e30f; l[v] = 0.f;
    int qr = qrow0 + v + 8 * half;
    csq[v] = (qr > 0) ? (float)csb[qr - 1] : 0.f;
  }

  int nkb = qb * 4 + 4;   // causal: k up to qb*128+127
  for (int kb = 0; kb < nkb; ++kb) {
    __syncthreads();
#if HAVE_TDM
    {
      size_t gbase = (((size_t)(b * NKVH_ + kvh)) * S_ + (size_t)kb * 32) * HD_;
      if (w == 0) tdm_load_2d(ks_lds, Kg + gbase, 128u, 32u, 128ull, KPAD_);
      if (w == 1) tdm_load_2d(vs_lds, Vg + gbase, 128u, 32u, 128ull, KPAD_);
      if (w < 2) __builtin_amdgcn_s_wait_tensorcnt(0);
    }
#else
    for (int i = tid; i < 32 * HD_; i += 256) {
      int r = i >> 7, d = i & 127;
      size_t gi = (((size_t)(b * NKVH_ + kvh)) * S_ + kb * 32 + r) * HD_ + d;
      Ks[r][d] = Kg[gi];
      Vs[r][d] = Vg[gi];
    }
#endif
    __syncthreads();

    // scores: S[q, kb*32 .. +31], two 16x16 C tiles, 4 d-chunks each
    v8f s0 = {}, s1 = {};
#pragma unroll
    for (int c = 0; c < 4; ++c) {
      v16bf b0, b1;
#pragma unroll
      for (int e = 0; e < 16; ++e) {
        b0[e] = Ks[ln][c * 32 + amap(e, half)];
        b1[e] = Ks[16 + ln][c * 32 + amap(e, half)];
      }
      s0 = wmma_bf16(qf[c], b0, s0);
      s1 = wmma_bf16(qf[c], b1, s1);
    }

    int kp0 = kb * 32 + ln, kp1 = kp0 + 16;
    int e0 = kp0 + 10; if (e0 > S_ - 1) e0 = S_ - 1;
    int e1 = kp1 + 10; if (e1 > S_ - 1) e1 = S_ - 1;
    float csb0 = (float)csb[e0];
    float csb1 = (float)csb[e1];

#pragma unroll
    for (int v = 0; v < 8; ++v) {
      int qr = qrow0 + v + 8 * half;
      float a0 = s0[v] * SCALE_;
      float a1 = s1[v] * SCALE_;
      if (kp0 > qr) a0 = -1e9f;
      else if (e0 >= qr && (csb0 - csq[v]) > 0.5f) a0 += 2.0f;
      if (kp1 > qr) a1 = -1e9f;
      else if (e1 >= qr && (csb1 - csq[v]) > 0.5f) a1 += 2.0f;

      float mx = fmaxf(a0, a1);
#pragma unroll
      for (int o = 8; o; o >>= 1) mx = fmaxf(mx, __shfl_xor(mx, o, 32));
      float mn = fmaxf(m[v], mx);
      float sc = __expf(m[v] - mn);
      float p0 = __expf(a0 - mn);
      float p1 = __expf(a1 - mn);
      float ps = p0 + p1;
#pragma unroll
      for (int o = 8; o; o >>= 1) ps += __shfl_xor(ps, o, 32);
      l[v] = l[v] * sc + ps;
      m[v] = mn;
#pragma unroll
      for (int dt = 0; dt < 8; ++dt) acc[dt][v] *= sc;

      Ps[w][v + 8 * half][ln]      = tobf(p0);
      Ps[w][v + 8 * half][16 + ln] = tobf(p1);
    }

    // per-wave LDS region: wait for our DS stores, then re-read in A layout
    asm volatile("s_wait_dscnt 0" ::: "memory");
    v16bf pf;
#pragma unroll
    for (int e = 0; e < 16; ++e) pf[e] = Ps[w][ln][amap(e, half)];

#pragma unroll
    for (int dt = 0; dt < 8; ++dt) {
      v16bf vb;
#pragma unroll
      for (int e = 0; e < 16; ++e) vb[e] = Vs[amap(e, half)][dt * 16 + ln];
      acc[dt] = wmma_bf16(pf, vb, acc[dt]);
    }
  }

#pragma unroll
  for (int dt = 0; dt < 8; ++dt)
#pragma unroll
    for (int v = 0; v < 8; ++v) {
      int qr = qrow0 + v + 8 * half;
      out[(size_t)(b * S_ + qr) * DIM_ + h * HD_ + dt * 16 + ln] = acc[dt][v] / l[v];
    }
}

// --------------------------------------------------------------- launch -----
extern "C" void kernel_launch(void* const* d_in, const int* in_sizes, int n_in,
                              void* d_out, int out_size, void* d_ws, size_t ws_size,
                              hipStream_t stream) {
  (void)in_sizes; (void)n_in; (void)out_size; (void)ws_size;
  const float* x      = (const float*)d_in[0];
  const float* freqs  = (const float*)d_in[2];
  const int*   labels = (const int*)d_in[4];
  const float* wq     = (const float*)d_in[5];
  const float* wk     = (const float*)d_in[6];
  const float* wv     = (const float*)d_in[7];
  const float* wo     = (const float*)d_in[8];
  const float* ln1w   = (const float*)d_in[9];
  const float* ln1b   = (const float*)d_in[10];
  const float* ln2w   = (const float*)d_in[11];
  const float* ln2b   = (const float*)d_in[12];

  constexpr size_t MB   = 1024 * 1024;
  constexpr size_t O_CS = 0;                       // 16 KB cumsum
  constexpr size_t O_XN = 64 * 1024;               // 16 MB bf16 LN output (reused for LN2)
  constexpr size_t O_QF = O_XN + 16 * MB;          // 32 MB f32 Q proj (reused as attn out)
  constexpr size_t O_KF = O_QF + 32 * MB;          //  8 MB f32 K proj
  constexpr size_t O_VF = O_KF + 8 * MB;           //  8 MB f32 V proj
  constexpr size_t O_QB = O_VF + 8 * MB;           // 16 MB bf16 Q packed
  constexpr size_t O_KB = O_QB + 16 * MB;          //  4 MB bf16 K packed
  constexpr size_t O_VB = O_KB + 4 * MB;           //  4 MB bf16 V packed

  char* ws = (char*)d_ws;
  int*    cs   = (int*)(ws + O_CS);
  __bf16* xn   = (__bf16*)(ws + O_XN);
  float*  qf   = (float*)(ws + O_QF);
  float*  kf   = (float*)(ws + O_KF);
  float*  vf   = (float*)(ws + O_VF);
  __bf16* qbuf = (__bf16*)(ws + O_QB);
  __bf16* kbuf = (__bf16*)(ws + O_KB);
  __bf16* vbuf = (__bf16*)(ws + O_VB);
  float*  attn = qf;   // reuse Q-proj buffer after packing

  k_cumsum<<<B_, 32, 0, stream>>>(labels, cs);
  k_ln<<<BS_, 256, 0, stream>>>(x, ln1w, ln1b, xn);

  k_gemm<<<dim3(DIM_ / 128, BS_ / 128), 256, 0, stream>>>(xn, wq, qf, DIM_, DIM_);
  k_gemm<<<dim3(512 / 128, BS_ / 128), 256, 0, stream>>>(xn, wk, kf, 512, DIM_);
  k_gemm<<<dim3(512 / 128, BS_ / 128), 256, 0, stream>>>(xn, wv, vf, 512, DIM_);

  k_rope<<<(BS_ * NH_ * 64) / 256, 256, 0, stream>>>(qf, freqs, qbuf, NH_, 1);
  k_rope<<<(BS_ * NKVH_ * 64) / 256, 256, 0, stream>>>(kf, freqs, kbuf, NKVH_, 1);
  k_rope<<<(BS_ * NKVH_ * 64) / 256, 256, 0, stream>>>(vf, freqs, vbuf, NKVH_, 0);

  k_attn<<<dim3(S_ / 128, B_ * NH_), 256, 0, stream>>>(qbuf, kbuf, vbuf, cs, attn);

  k_ln<<<BS_, 256, 0, stream>>>(attn, ln2w, ln2b, xn);
  k_gemm<<<dim3(DIM_ / 128, BS_ / 128), 256, 0, stream>>>(xn, wo, (float*)d_out, DIM_, DIM_);
}
